// GraphTransformerLayer_68461778698591
// MI455X (gfx1250) — compile-verified, matrix operand order checked by reading, not run on hardware
//
#include <hip/hip_runtime.h>

// ---------------------------------------------------------------------------
// Graph Transformer layer for MI455X (gfx1250, wave32, WMMA).
// All 128-deep GEMMs run on v_wmma_f32_16x16x32_bf16 (fp32 accumulate).
// Weights are converted once per launch to transposed bf16 in workspace.
// ---------------------------------------------------------------------------

constexpr int DM  = 128;      // model dim
constexpr int NN  = 50000;    // nodes
constexpr int NE  = 400000;   // edges
constexpr int RPB = 64;       // rows per GEMM block (4 waves x 16 rows)

typedef __attribute__((ext_vector_type(16))) __bf16       v16bf;
typedef __attribute__((ext_vector_type(8)))  float        v8f;
typedef __attribute__((ext_vector_type(4)))  unsigned int v4u;

struct FragU { v4u lo; v4u hi; };
static_assert(sizeof(FragU) == sizeof(v16bf), "frag size mismatch");

// float -> bf16 round-to-nearest-even
__device__ __forceinline__ unsigned short f2bf(float x) {
  unsigned int u = __builtin_bit_cast(unsigned int, x);
  u += 0x7fffu + ((u >> 16) & 1u);
  return (unsigned short)(u >> 16);
}

// Build a 16x32 bf16 fragment (v16bf, 8 VGPRs) from an LDS row-major tile.
// ISA layout: lanes 0-15 hold K = {k0..k0+7, k0+16..k0+23},
//             lanes 16-31 hold K = {k0+8..k0+15, k0+24..k0+31} (half = lane>>4).
__device__ __forceinline__ v16bf lds_frag(const unsigned short* base, int row,
                                          int k0, int half) {
  const v4u* p = (const v4u*)(base + row * DM + k0 + half * 8);
  FragU f;
  f.lo = p[0];   // 8 bf16 @ k0 + half*8
  f.hi = p[2];   // 8 bf16 @ k0 + 16 + half*8
  return __builtin_bit_cast(v16bf, f);
}

__device__ __forceinline__ v8f wmma_bf16(v16bf a, v16bf b, v8f c) {
  return __builtin_amdgcn_wmma_f32_16x16x32_bf16(false, a, false, b,
                                                 (short)0, c, false, false);
}

// Reduce sum/sumsq over the 16 lanes of a half-wave (row spread over 16 lanes).
__device__ __forceinline__ void stats16(float s, float s2, float& mu, float& rs) {
#pragma unroll
  for (int m = 1; m < 16; m <<= 1) {
    s  += __shfl_xor(s,  m, 32);
    s2 += __shfl_xor(s2, m, 32);
  }
  mu = s * (1.0f / DM);
  rs = rsqrtf(s2 * (1.0f / DM) - mu * mu + 1e-5f);
}

// Reduce over all 32 lanes (row spread as 32 lanes x 4 values).
__device__ __forceinline__ void stats32(float s, float s2, float& mu, float& rs) {
#pragma unroll
  for (int m = 1; m < 32; m <<= 1) {
    s  += __shfl_xor(s,  m, 32);
    s2 += __shfl_xor(s2, m, 32);
  }
  mu = s * (1.0f / DM);
  rs = rsqrtf(s2 * (1.0f / DM) - mu * mu + 1e-5f);
}

// Copy a 128x128 bf16 (transposed) weight matrix from global into LDS.
__device__ __forceinline__ void stage_w(unsigned short* ldsW,
                                        const unsigned short* WT, int tid) {
  const v4u* s = (const v4u*)WT;
  v4u* d = (v4u*)ldsW;
#pragma unroll 4
  for (int i = tid; i < 2048; i += 128) d[i] = s[i];
}

// One wave computes its 16-row strip times the staged 128x128 weight:
// acc[nt] (nt = 0..7) accumulates the 16x16 tile at columns nt*16.
__device__ __forceinline__ void gemm_row(const unsigned short* A,
                                         const unsigned short* W, int mBase,
                                         int idx, int half, v8f acc[8]) {
  v16bf af[4];
#pragma unroll
  for (int kk = 0; kk < 4; ++kk) af[kk] = lds_frag(A, mBase + idx, kk * 32, half);
#pragma unroll
  for (int nt = 0; nt < 8; ++nt) {
    v8f c = acc[nt];
#pragma unroll
    for (int kk = 0; kk < 4; ++kk) {
      v16bf b = lds_frag(W, nt * 16 + idx, kk * 32, half);
      c = wmma_bf16(af[kk], b, c);
    }
    acc[nt] = c;
  }
}

// Stage 16 rows of X (with LayerNorm) into LDS as bf16. One wave = 16 rows.
__device__ __forceinline__ void stage_rows_ln(unsigned short* ldsA,
                                              const float* X, const float* g,
                                              const float* b, int r0, int nrows,
                                              int wave, int lane) {
  for (int rr = 0; rr < 16; ++rr) {
    int lr = wave * 16 + rr;
    int row = r0 + lr;
    float v[4];
    float s = 0.f, s2 = 0.f;
    if (row < nrows) {
#pragma unroll
      for (int i = 0; i < 4; ++i) {
        v[i] = X[(size_t)row * DM + lane * 4 + i];
        s += v[i];
        s2 += v[i] * v[i];
      }
    } else {
      v[0] = v[1] = v[2] = v[3] = 0.f;
    }
    float mu, rs;
    stats32(s, s2, mu, rs);
#pragma unroll
    for (int i = 0; i < 4; ++i) {
      int c = lane * 4 + i;
      float y = (row < nrows) ? ((v[i] - mu) * rs * g[c] + b[c]) : 0.f;
      ldsA[lr * DM + c] = f2bf(y);
    }
  }
}

// Epilogue: y = LN(acc + bias [+ bias2]) with ln params (g,b); store or +=.
__device__ __forceinline__ void ln_epilogue_store(v8f acc[8], const float* bias,
                                                  const float* bias2,
                                                  const float* g, const float* b,
                                                  float* out, int r0, int wave,
                                                  int idx, int half, int nrows,
                                                  bool accumulate) {
#pragma unroll
  for (int j = 0; j < 8; ++j) {
    int row = r0 + wave * 16 + half * 8 + j;
    float tv[8];
    float s = 0.f, s2 = 0.f;
#pragma unroll
    for (int nt = 0; nt < 8; ++nt) {
      int c = nt * 16 + idx;
      float x = acc[nt][j] + bias[c] + (bias2 ? bias2[c] : 0.f);
      tv[nt] = x;
      s += x;
      s2 += x * x;
    }
    float mu, rs;
    stats16(s, s2, mu, rs);
    if (row < nrows) {
#pragma unroll
      for (int nt = 0; nt < 8; ++nt) {
        int c = nt * 16 + idx;
        float y = (tv[nt] - mu) * rs * g[c] + b[c];
        if (accumulate)
          out[(size_t)row * DM + c] += y;
        else
          out[(size_t)row * DM + c] = y;
      }
    }
  }
}

// ---------------------------------------------------------------------------
// Kernel: convert fp32 weight (din x dout, row-major) -> transposed bf16.
// ---------------------------------------------------------------------------
__global__ void k_w2bf(const float* __restrict__ W, unsigned short* __restrict__ WT,
                       int din, int dout) {
  int i = blockIdx.x * blockDim.x + threadIdx.x;
  if (i < din * dout) {
    int k = i / dout, n = i % dout;
    WT[(size_t)n * din + k] = f2bf(W[i]);
  }
}

__global__ void k_zero(float* __restrict__ p, long n) {
  long i = (long)blockIdx.x * blockDim.x + threadIdx.x;
  long stride = (long)gridDim.x * blockDim.x;
  for (; i < n; i += stride) p[i] = 0.f;
}

// ---------------------------------------------------------------------------
// Kernel: Q/K/V projections.  A = bf16(LN(h, ln1_h)); for W in {Q,K,V}:
//   out = LN(A @ W + b, ln_attn)
// ---------------------------------------------------------------------------
__global__ __launch_bounds__(128) void k_qkv(
    const float* __restrict__ h, int nrows, const float* __restrict__ lng,
    const float* __restrict__ lnb, const unsigned short* __restrict__ WTQ,
    const float* __restrict__ bQ, const unsigned short* __restrict__ WTK,
    const float* __restrict__ bK, const unsigned short* __restrict__ WTV,
    const float* __restrict__ bV, const float* __restrict__ lag,
    const float* __restrict__ lab, float* __restrict__ Qh,
    float* __restrict__ Kh, float* __restrict__ Vh) {
  __shared__ unsigned short ldsA[RPB * DM];
  __shared__ unsigned short ldsW[DM * DM];
  int tid = threadIdx.x, wave = tid >> 5, lane = tid & 31;
  int half = lane >> 4, idx = lane & 15;
  int r0 = blockIdx.x * RPB;

  stage_rows_ln(ldsA, h, lng, lnb, r0, nrows, wave, lane);

  const unsigned short* Ws[3] = {WTQ, WTK, WTV};
  const float* Bs[3] = {bQ, bK, bV};
  float* Os[3] = {Qh, Kh, Vh};

#pragma unroll
  for (int m = 0; m < 3; ++m) {
    __syncthreads();
    stage_w(ldsW, Ws[m], tid);
    __syncthreads();
    v8f acc[8];
    v8f vz = {};
#pragma unroll
    for (int nt = 0; nt < 8; ++nt) acc[nt] = vz;
    gemm_row(ldsA, ldsW, wave * 16, idx, half, acc);
    ln_epilogue_store(acc, Bs[m], nullptr, lag, lab, Os[m], r0, wave, idx, half,
                      nrows, false);
  }
}

// ---------------------------------------------------------------------------
// Kernel: edge position/feature projections.
//   P  = LN(LN(e,ln1_e) @ Wpe + bpe, ln_attn)
//      + LN(spatial @ Wpos + bpos + pos_emb, ln_attn)
// ---------------------------------------------------------------------------
__global__ __launch_bounds__(128) void k_edge_pp(
    const float* __restrict__ e, const float* __restrict__ sp, int nrows,
    const float* __restrict__ lng, const float* __restrict__ lnb,
    const unsigned short* __restrict__ WTpe, const float* __restrict__ bpe,
    const unsigned short* __restrict__ WTpos, const float* __restrict__ bpos,
    const float* __restrict__ pose, const float* __restrict__ lag,
    const float* __restrict__ lab, float* __restrict__ P) {
  __shared__ unsigned short ldsE[RPB * DM];
  __shared__ unsigned short ldsS[RPB * DM];
  __shared__ unsigned short ldsW[DM * DM];
  int tid = threadIdx.x, wave = tid >> 5, lane = tid & 31;
  int half = lane >> 4, idx = lane & 15;
  int r0 = blockIdx.x * RPB;

  // A_e = bf16(LN(e)), A_s = bf16(spatial) (no LN on spatial input)
  stage_rows_ln(ldsE, e, lng, lnb, r0, nrows, wave, lane);
  for (int rr = 0; rr < 16; ++rr) {
    int lr = wave * 16 + rr;
    int row = r0 + lr;
#pragma unroll
    for (int i = 0; i < 4; ++i) {
      int c = lane * 4 + i;
      float v = (row < nrows) ? sp[(size_t)row * DM + c] : 0.f;
      ldsS[lr * DM + c] = f2bf(v);
    }
  }

  v8f vz = {};
  // Phase 1: proj_e
  __syncthreads();
  stage_w(ldsW, WTpe, tid);
  __syncthreads();
  {
    v8f acc[8];
#pragma unroll
    for (int nt = 0; nt < 8; ++nt) acc[nt] = vz;
    gemm_row(ldsE, ldsW, wave * 16, idx, half, acc);
    ln_epilogue_store(acc, bpe, nullptr, lag, lab, P, r0, wave, idx, half, nrows,
                      false);
  }
  // Phase 2: positional encoding (+pos_emb folded into bias)
  __syncthreads();
  stage_w(ldsW, WTpos, tid);
  __syncthreads();
  {
    v8f acc[8];
#pragma unroll
    for (int nt = 0; nt < 8; ++nt) acc[nt] = vz;
    gemm_row(ldsS, ldsW, wave * 16, idx, half, acc);
    ln_epilogue_store(acc, bpos, pose, lag, lab, P, r0, wave, idx, half, nrows,
                      true);
  }
}

// ---------------------------------------------------------------------------
// Kernel: edge attention. One wave per edge (32 lanes x 4 channels = 128).
// score = clip(K[src]*Q[dst]/(4*T) + P, +-5); softmax per 16-ch head;
// P <- score; wV[dst] += V[src]*score; z[dst] += score  (f32 atomics).
// ---------------------------------------------------------------------------
__global__ __launch_bounds__(256) void k_attn(
    const float* __restrict__ Qh, const float* __restrict__ Kh,
    const float* __restrict__ Vh, float* __restrict__ P,
    const int* __restrict__ src, const int* __restrict__ dst,
    const float* __restrict__ temp, float* __restrict__ wV,
    float* __restrict__ z, int nedges) {
  int gw = (int)((blockIdx.x * blockDim.x + threadIdx.x) >> 5);
  int lane = threadIdx.x & 31;
  int nw = (int)((gridDim.x * blockDim.x) >> 5);
  float inv = 1.0f / (4.0f * temp[0]);

  for (int eidx = gw; eidx < nedges; eidx += nw) {
    int s = src[eidx], d = dst[eidx];
    float4 kk = ((const float4*)(Kh + (size_t)s * DM))[lane];
    float4 qq = ((const float4*)(Qh + (size_t)d * DM))[lane];
    float4 vv = ((const float4*)(Vh + (size_t)s * DM))[lane];
    float4 pp = ((const float4*)(P + (size_t)eidx * DM))[lane];
    float sc[4];
    sc[0] = kk.x * qq.x * inv + pp.x;
    sc[1] = kk.y * qq.y * inv + pp.y;
    sc[2] = kk.z * qq.z * inv + pp.z;
    sc[3] = kk.w * qq.w * inv + pp.w;
#pragma unroll
    for (int i = 0; i < 4; ++i) sc[i] = fminf(5.f, fmaxf(-5.f, sc[i]));

    // softmax over each head: 16 contiguous channels = quad of lanes
    float mx = fmaxf(fmaxf(sc[0], sc[1]), fmaxf(sc[2], sc[3]));
    mx = fmaxf(mx, __shfl_xor(mx, 1, 32));
    mx = fmaxf(mx, __shfl_xor(mx, 2, 32));
    float sum = 0.f;
#pragma unroll
    for (int i = 0; i < 4; ++i) {
      sc[i] = __expf(sc[i] - mx);
      sum += sc[i];
    }
    sum += __shfl_xor(sum, 1, 32);
    sum += __shfl_xor(sum, 2, 32);
    float r = 1.0f / sum;

    float4 so;
    so.x = sc[0] * r; so.y = sc[1] * r; so.z = sc[2] * r; so.w = sc[3] * r;
    ((float4*)(P + (size_t)eidx * DM))[lane] = so;

    int c = lane * 4;
    float* wrow = wV + (size_t)d * DM + c;
    float* zrow = z + (size_t)d * DM + c;
    atomicAdd(wrow + 0, vv.x * so.x);
    atomicAdd(wrow + 1, vv.y * so.y);
    atomicAdd(wrow + 2, vv.z * so.z);
    atomicAdd(wrow + 3, vv.w * so.w);
    atomicAdd(zrow + 0, so.x);
    atomicAdd(zrow + 1, so.y);
    atomicAdd(zrow + 2, so.z);
    atomicAdd(zrow + 3, so.w);
  }
}

// ---------------------------------------------------------------------------
// Kernel: output projection + residual + BN + LN prep (nodes & edges).
//   A = wV/(z+1e-6)   (nodes)  or  scores (edges)
//   X2 = BN(resid + A @ Wo + bo);  XN = bf16(LN(X2, ln2))
// ---------------------------------------------------------------------------
__global__ __launch_bounds__(128) void k_post(
    const float* __restrict__ att, const float* __restrict__ wV,
    const float* __restrict__ z, int is_node, const float* __restrict__ resid,
    const unsigned short* __restrict__ WTo, const float* __restrict__ bo,
    const float* __restrict__ bg, const float* __restrict__ bb,
    const float* __restrict__ brm, const float* __restrict__ brv,
    const float* __restrict__ lg, const float* __restrict__ lb,
    float* __restrict__ X2, unsigned short* __restrict__ XN, int nrows) {
  __shared__ unsigned short ldsA[RPB * DM];
  __shared__ unsigned short ldsW[DM * DM];
  int tid = threadIdx.x, wave = tid >> 5, lane = tid & 31;
  int half = lane >> 4, idx = lane & 15;
  int r0 = blockIdx.x * RPB;

  for (int rr = 0; rr < 16; ++rr) {
    int lr = wave * 16 + rr;
    int row = r0 + lr;
#pragma unroll
    for (int i = 0; i < 4; ++i) {
      int c = lane * 4 + i;
      float a = 0.f;
      if (row < nrows) {
        if (is_node) {
          float zz = z[(size_t)row * DM + c];
          a = wV[(size_t)row * DM + c] / (zz + 1e-6f);
        } else {
          a = att[(size_t)row * DM + c];
        }
      }
      ldsA[lr * DM + c] = f2bf(a);
    }
  }
  __syncthreads();
  stage_w(ldsW, WTo, tid);
  __syncthreads();

  v8f acc[8];
  v8f vz = {};
#pragma unroll
  for (int nt = 0; nt < 8; ++nt) acc[nt] = vz;
  gemm_row(ldsA, ldsW, wave * 16, idx, half, acc);

#pragma unroll
  for (int j = 0; j < 8; ++j) {
    int row = r0 + wave * 16 + half * 8 + j;
    bool ok = row < nrows;
    float tv[8];
    float s = 0.f, s2 = 0.f;
#pragma unroll
    for (int nt = 0; nt < 8; ++nt) {
      int c = nt * 16 + idx;
      float x = acc[nt][j] + bo[c] + (ok ? resid[(size_t)row * DM + c] : 0.f);
      float t = (x - brm[c]) * rsqrtf(brv[c] + 1e-5f) * bg[c] + bb[c];
      tv[nt] = t;
      s += t;
      s2 += t * t;
    }
    float mu, rs;
    stats16(s, s2, mu, rs);
    if (ok) {
#pragma unroll
      for (int nt = 0; nt < 8; ++nt) {
        int c = nt * 16 + idx;
        X2[(size_t)row * DM + c] = tv[nt];
        XN[(size_t)row * DM + c] = f2bf((tv[nt] - mu) * rs * lg[c] + lb[c]);
      }
    }
  }
}

// ---------------------------------------------------------------------------
// Kernel: fused FFN 128 -> 512 -> 128 (relu), residual, BN2.
//   out = BN2(X2 + relu(XN @ W1 + b1) @ W2 + b2)
// Hidden dim processed in 4 chunks of 128; intermediate stays in LDS as bf16.
// ---------------------------------------------------------------------------
__global__ __launch_bounds__(128) void k_ffn(
    const unsigned short* __restrict__ XN, const float* __restrict__ X2,
    const unsigned short* __restrict__ WT1, const float* __restrict__ b1,
    const unsigned short* __restrict__ WT2, const float* __restrict__ b2,
    const float* __restrict__ bg, const float* __restrict__ bb,
    const float* __restrict__ brm, const float* __restrict__ brv,
    float* __restrict__ out, int nrows) {
  __shared__ unsigned short ldsA[RPB * DM];
  __shared__ unsigned short ldsW[DM * DM];
  __shared__ unsigned short ldsT[RPB * DM];
  int tid = threadIdx.x, wave = tid >> 5, lane = tid & 31;
  int half = lane >> 4, idx = lane & 15;
  int r0 = blockIdx.x * RPB;
  (void)lane;

  // stage A (already bf16)
  {
    v4u zero4 = {};
    v4u* d = (v4u*)ldsA;
    for (int i = tid; i < 1024; i += 128) {
      int lr = i >> 4;
      int row = r0 + lr;
      d[i] = (row < nrows) ? ((const v4u*)XN)[(size_t)row * 16 + (i & 15)] : zero4;
    }
  }

  v8f acc[8];
  v8f vz = {};
#pragma unroll
  for (int nt = 0; nt < 8; ++nt) acc[nt] = vz;

  for (int hc = 0; hc < 512; hc += 128) {
    __syncthreads();
    stage_w(ldsW, WT1 + (size_t)hc * DM, tid);
    if (hc + 128 < 512) __builtin_prefetch(WT1 + (size_t)(hc + 128) * DM, 0, 1);
    __syncthreads();

    // T = bf16(relu(A @ W1chunk + b1))
#pragma unroll
    for (int nt = 0; nt < 8; ++nt) {
      v8f t = vz;
      v16bf af[4];
#pragma unroll
      for (int kk = 0; kk < 4; ++kk)
        af[kk] = lds_frag(ldsA, wave * 16 + idx, kk * 32, half);
#pragma unroll
      for (int kk = 0; kk < 4; ++kk) {
        v16bf b = lds_frag(ldsW, nt * 16 + idx, kk * 32, half);
        t = wmma_bf16(af[kk], b, t);
      }
#pragma unroll
      for (int j = 0; j < 8; ++j) {
        int lr = wave * 16 + half * 8 + j;
        int c = nt * 16 + idx;
        float y = t[j] + b1[hc + c];
        ldsT[lr * DM + c] = f2bf(fmaxf(y, 0.f));
      }
    }
    __syncthreads();

    // stage W2 chunk: ldsW[n][k'] = WT2[n*512 + hc + k']
    {
      v4u* d = (v4u*)ldsW;
      for (int i = tid; i < 2048; i += 128) {
        int n = i >> 4;
        int kq = i & 15;
        d[i] = *(const v4u*)(WT2 + (size_t)n * 512 + hc + kq * 8);
      }
    }
    __syncthreads();

    // acc += T @ W2chunk
    gemm_row(ldsT, ldsW, wave * 16, idx, half, acc);
  }

  // epilogue: residual + bias + BN2 -> out
#pragma unroll
  for (int nt = 0; nt < 8; ++nt) {
#pragma unroll
    for (int j = 0; j < 8; ++j) {
      int row = r0 + wave * 16 + half * 8 + j;
      int c = nt * 16 + idx;
      if (row < nrows) {
        float y = acc[nt][j] + b2[c] + X2[(size_t)row * DM + c];
        y = (y - brm[c]) * rsqrtf(brv[c] + 1e-5f) * bg[c] + bb[c];
        out[(size_t)row * DM + c] = y;
      }
    }
  }
}

// ---------------------------------------------------------------------------
// Host orchestration
// ---------------------------------------------------------------------------
extern "C" void kernel_launch(void* const* d_in, const int* in_sizes, int n_in,
                              void* d_out, int out_size, void* d_ws,
                              size_t ws_size, hipStream_t stream) {
  (void)in_sizes; (void)n_in; (void)out_size; (void)ws_size;

  // ---- inputs (setup_inputs order, params dict flattened in insertion order)
  const float* h   = (const float*)d_in[0];
  const float* e   = (const float*)d_in[1];
  const float* sp  = (const float*)d_in[2];
  const int* src   = (const int*)d_in[3];
  const int* dst   = (const int*)d_in[4];
  // d_in[5] = n_nodes (scalar) -- compile-time constants used instead
  const float* QW  = (const float*)d_in[6];  const float* Qb  = (const float*)d_in[7];
  const float* KW  = (const float*)d_in[8];  const float* Kb  = (const float*)d_in[9];
  const float* VW  = (const float*)d_in[10]; const float* Vb  = (const float*)d_in[11];
  const float* PEW = (const float*)d_in[12]; const float* PEb = (const float*)d_in[13];
  const float* POW = (const float*)d_in[14]; const float* POb = (const float*)d_in[15];
  const float* pose = (const float*)d_in[16];
  const float* temp = (const float*)d_in[17];
  const float* lag = (const float*)d_in[18]; const float* lab = (const float*)d_in[19];
  const float* OHW = (const float*)d_in[20]; const float* OHb = (const float*)d_in[21];
  const float* OEW = (const float*)d_in[22]; const float* OEb = (const float*)d_in[23];
  const float* l1hg = (const float*)d_in[24]; const float* l1hb = (const float*)d_in[25];
  const float* l1eg = (const float*)d_in[26]; const float* l1eb = (const float*)d_in[27];
  const float* l2hg = (const float*)d_in[28]; const float* l2hb = (const float*)d_in[29];
  const float* l2eg = (const float*)d_in[30]; const float* l2eb = (const float*)d_in[31];
  const float* b1hg = (const float*)d_in[32]; const float* b1hb = (const float*)d_in[33];
  const float* b1hm = (const float*)d_in[34]; const float* b1hv = (const float*)d_in[35];
  const float* b1eg = (const float*)d_in[36]; const float* b1eb = (const float*)d_in[37];
  const float* b1em = (const float*)d_in[38]; const float* b1ev = (const float*)d_in[39];
  const float* b2hg = (const float*)d_in[40]; const float* b2hb = (const float*)d_in[41];
  const float* b2hm = (const float*)d_in[42]; const float* b2hv = (const float*)d_in[43];
  const float* b2eg = (const float*)d_in[44]; const float* b2eb = (const float*)d_in[45];
  const float* b2em = (const float*)d_in[46]; const float* b2ev = (const float*)d_in[47];
  const float* F1hW = (const float*)d_in[48]; const float* F1hb = (const float*)d_in[49];
  const float* F2hW = (const float*)d_in[50]; const float* F2hb = (const float*)d_in[51];
  const float* F1eW = (const float*)d_in[52]; const float* F1eb = (const float*)d_in[53];
  const float* F2eW = (const float*)d_in[54]; const float* F2eb = (const float*)d_in[55];

  // ---- workspace carve-up
  char* ws = (char*)d_ws;
  size_t off = 0;
  auto take = [&](size_t bytes) -> void* {
    void* p = ws + off;
    off = (off + bytes + 255) & ~(size_t)255;
    return p;
  };
  unsigned short* tQ  = (unsigned short*)take(DM * DM * 2);
  unsigned short* tK  = (unsigned short*)take(DM * DM * 2);
  unsigned short* tV  = (unsigned short*)take(DM * DM * 2);
  unsigned short* tPE = (unsigned short*)take(DM * DM * 2);
  unsigned short* tPO = (unsigned short*)take(DM * DM * 2);
  unsigned short* tOH = (unsigned short*)take(DM * DM * 2);
  unsigned short* tOE = (unsigned short*)take(DM * DM * 2);
  unsigned short* t1H = (unsigned short*)take((size_t)DM * 512 * 2);
  unsigned short* t2H = (unsigned short*)take((size_t)DM * 512 * 2);
  unsigned short* t1E = (unsigned short*)take((size_t)DM * 512 * 2);
  unsigned short* t2E = (unsigned short*)take((size_t)DM * 512 * 2);
  float* Qh = (float*)take((size_t)NN * DM * 4);
  float* Kh = (float*)take((size_t)NN * DM * 4);
  float* Vh = (float*)take((size_t)NN * DM * 4);
  float* P  = (float*)take((size_t)NE * DM * 4);   // edge proj+pos, then scores
  float* wV = (float*)take((size_t)NN * DM * 4);
  float* zb = (float*)take((size_t)NN * DM * 4);
  float* H2 = (float*)take((size_t)NN * DM * 4);
  unsigned short* HN = (unsigned short*)take((size_t)NN * DM * 2);
  float* E2 = (float*)take((size_t)NE * DM * 4);
  unsigned short* EN = (unsigned short*)take((size_t)NE * DM * 2);

  float* outH = (float*)d_out;
  float* outE = (float*)d_out + (size_t)NN * DM;

  // ---- weight conversion (fp32 -> transposed bf16)
  k_w2bf<<<64, 256, 0, stream>>>(QW,  tQ,  DM, DM);
  k_w2bf<<<64, 256, 0, stream>>>(KW,  tK,  DM, DM);
  k_w2bf<<<64, 256, 0, stream>>>(VW,  tV,  DM, DM);
  k_w2bf<<<64, 256, 0, stream>>>(PEW, tPE, DM, DM);
  k_w2bf<<<64, 256, 0, stream>>>(POW, tPO, DM, DM);
  k_w2bf<<<64, 256, 0, stream>>>(OHW, tOH, DM, DM);
  k_w2bf<<<64, 256, 0, stream>>>(OEW, tOE, DM, DM);
  k_w2bf<<<256, 256, 0, stream>>>(F1hW, t1H, DM, 512);
  k_w2bf<<<256, 256, 0, stream>>>(F2hW, t2H, 512, DM);
  k_w2bf<<<256, 256, 0, stream>>>(F1eW, t1E, DM, 512);
  k_w2bf<<<256, 256, 0, stream>>>(F2eW, t2E, 512, DM);

  // ---- zero accumulators
  k_zero<<<4096, 256, 0, stream>>>(wV, (long)NN * DM);
  k_zero<<<4096, 256, 0, stream>>>(zb, (long)NN * DM);

  const int nodeBlocks = (NN + RPB - 1) / RPB;   // 782
  const int edgeBlocks = (NE + RPB - 1) / RPB;   // 6250

  // ---- attention front-end
  k_qkv<<<nodeBlocks, 128, 0, stream>>>(h, NN, l1hg, l1hb, tQ, Qb, tK, Kb, tV,
                                        Vb, lag, lab, Qh, Kh, Vh);
  k_edge_pp<<<edgeBlocks, 128, 0, stream>>>(e, sp, NE, l1eg, l1eb, tPE, PEb,
                                            tPO, POb, pose, lag, lab, P);
  k_attn<<<12500, 256, 0, stream>>>(Qh, Kh, Vh, P, src, dst, temp, wV, zb, NE);

  // ---- output projection + BN1 + LN2 prep
  k_post<<<nodeBlocks, 128, 0, stream>>>(nullptr, wV, zb, 1, h, tOH, OHb, b1hg,
                                         b1hb, b1hm, b1hv, l2hg, l2hb, H2, HN,
                                         NN);
  k_post<<<edgeBlocks, 128, 0, stream>>>(P, nullptr, nullptr, 0, e, tOE, OEb,
                                         b1eg, b1eb, b1em, b1ev, l2eg, l2eb, E2,
                                         EN, NE);

  // ---- FFN + BN2 -> outputs
  k_ffn<<<nodeBlocks, 128, 0, stream>>>(HN, H2, t1H, F1hb, t2H, F2hb, b2hg,
                                        b2hb, b2hm, b2hv, outH, NN);
  k_ffn<<<edgeBlocks, 128, 0, stream>>>(EN, E2, t1E, F1eb, t2E, F2eb, b2eg,
                                        b2eb, b2em, b2ev, outE, NE);
}
